// CRELayerBase_37374805410135
// MI455X (gfx1250) — matrix operand order repaired and checked by reading, not compiled, not run on hardware
//
#include <hip/hip_runtime.h>
#include <hip/hip_bf16.h>
#include <math.h>

#define D_DIM 1024
#define T_DIM 4096
#define B_DIM 4
#define M_DIM (B_DIM * T_DIM)   // 16384
#define LN_EPS 1e-5f

typedef float  v8f     __attribute__((ext_vector_type(8)));
typedef float  f32x4   __attribute__((ext_vector_type(4)));
typedef __bf16 bf16x16 __attribute__((ext_vector_type(16)));
typedef __bf16 bf16x8  __attribute__((ext_vector_type(8)));
typedef __bf16 bf16x4  __attribute__((ext_vector_type(4)));

union Frag { bf16x16 v; bf16x8 h[2]; };

static __device__ __forceinline__ __bf16 f2bf(float f) {
  unsigned u = __builtin_bit_cast(unsigned, f);
  unsigned r = u + 0x7FFFu + ((u >> 16) & 1u);   // round-to-nearest-even
  unsigned short h = (unsigned short)(r >> 16);
  return __builtin_bit_cast(__bf16, h);
}

// ---------------------------------------------------------------- converts
__global__ void cvt_f32_bf16_kernel(const float* __restrict__ src,
                                    __bf16* __restrict__ dst, int n) {
  int i = blockIdx.x * blockDim.x + threadIdx.x;
  int stride = gridDim.x * blockDim.x;
  for (; i < n; i += stride) dst[i] = f2bf(src[i]);
}

// ---------------------------------------------------------------- layernorm
__global__ __launch_bounds__(256) void ln_bf16_kernel(
    const float* __restrict__ x, const float* __restrict__ w,
    const float* __restrict__ b, __bf16* __restrict__ out) {
  const int row = blockIdx.x;
  const int tid = threadIdx.x;
  const float* xr = x + (size_t)row * D_DIM;
  f32x4 v = *(const f32x4*)(xr + tid * 4);
  float s  = v.x + v.y + v.z + v.w;
  float sq = v.x * v.x + v.y * v.y + v.z * v.z + v.w * v.w;
  #pragma unroll
  for (int off = 16; off >= 1; off >>= 1) {
    s  += __shfl_xor(s,  off, 32);
    sq += __shfl_xor(sq, off, 32);
  }
  __shared__ float ss[8], sqs[8];
  if ((tid & 31) == 0) { ss[tid >> 5] = s; sqs[tid >> 5] = sq; }
  __syncthreads();
  float ts = 0.f, tq = 0.f;
  #pragma unroll
  for (int i = 0; i < 8; i++) { ts += ss[i]; tq += sqs[i]; }
  float mu  = ts * (1.0f / D_DIM);
  float var = tq * (1.0f / D_DIM) - mu * mu;
  float rs  = rsqrtf(var + LN_EPS);
  f32x4 wv = *(const f32x4*)(w + tid * 4);
  f32x4 bv = *(const f32x4*)(b + tid * 4);
  bf16x4 o;
  o.x = f2bf((v.x - mu) * rs * wv.x + bv.x);
  o.y = f2bf((v.y - mu) * rs * wv.y + bv.y);
  o.z = f2bf((v.z - mu) * rs * wv.z + bv.z);
  o.w = f2bf((v.w - mu) * rs * wv.w + bv.w);
  *(bf16x4*)(out + (size_t)row * D_DIM + tid * 4) = o;
}

// ---------------------------------------------------------------- GEMM: gate+value fused
// C[m,n] = sum_k A[m,k] * W[n,k]   (torch Linear: x @ W.T)
// Workgroup tile 128(M) x 64(N), 8 waves, wave tile 32x32, K-step 32.
__global__ __launch_bounds__(256) void gemm_gv_kernel(
    const __bf16* __restrict__ A, const __bf16* __restrict__ Wg,
    const __bf16* __restrict__ Wv, const float* __restrict__ gb,
    const float* __restrict__ vb, float* __restrict__ U) {
  __shared__ __align__(16) __bf16 As[128 * 40];
  __shared__ __align__(16) __bf16 Bg[64 * 40];
  __shared__ __align__(16) __bf16 Bv[64 * 40];
  const int tid  = threadIdx.x;
  const int lane = tid & 31, wave = tid >> 5;
  const int wm = wave & 3, wn = wave >> 2;
  const int tileM = blockIdx.x * 128;
  const int tileN = blockIdx.y * 64;
  const int fr = lane & 15;
  const int half = lane >> 4;

  // per-thread staging coordinates
  const int ar0 = tid >> 2,        ac0 = (tid & 3) * 8;            // A chunk 0
  const int ar1 = (tid + 256) >> 2, ac1 = ((tid + 256) & 3) * 8;   // A chunk 1
  const int br  = tid >> 2,        bc  = (tid & 3) * 8;            // B chunk

  v8f accG[2][2] = {};
  v8f accV[2][2] = {};

  for (int k0 = 0; k0 < D_DIM; k0 += 32) {
    // 1) issue ALL global loads first so they can be clause-batched
    bf16x8 tA0 = *(const bf16x8*)&A [(size_t)(tileM + ar0) * D_DIM + k0 + ac0];
    bf16x8 tA1 = *(const bf16x8*)&A [(size_t)(tileM + ar1) * D_DIM + k0 + ac1];
    bf16x8 tBg = *(const bf16x8*)&Wg[(size_t)(tileN + br ) * D_DIM + k0 + bc ];
    bf16x8 tBv = *(const bf16x8*)&Wv[(size_t)(tileN + br ) * D_DIM + k0 + bc ];
    // 2) L2 prefetch of A two K-steps ahead (weights already L2-resident)
    if (k0 + 96 < D_DIM) {
      __builtin_prefetch(&A[(size_t)(tileM + ar0) * D_DIM + k0 + 64 + ac0], 0, 1);
      __builtin_prefetch(&A[(size_t)(tileM + ar1) * D_DIM + k0 + 64 + ac1], 0, 1);
    }
    // 3) stage into padded LDS (row stride 40 -> conflict-free b128 frag reads)
    *(bf16x8*)&As[ar0 * 40 + ac0] = tA0;
    *(bf16x8*)&As[ar1 * 40 + ac1] = tA1;
    *(bf16x8*)&Bg[br  * 40 + bc ] = tBg;
    *(bf16x8*)&Bv[br  * 40 + bc ] = tBv;
    __syncthreads();

    Frag a[2], bg[2], bv[2];
    #pragma unroll
    for (int i = 0; i < 2; i++) {            // A frag: K {0-7,16-23} / {8-15,24-31}
      int row = wm * 32 + i * 16 + fr;
      a[i].h[0] = *(const bf16x8*)&As[row * 40 + half * 8];
      a[i].h[1] = *(const bf16x8*)&As[row * 40 + half * 8 + 16];
    }
    #pragma unroll
    for (int j = 0; j < 2; j++) {            // B frag: K {0-7,16-23} / {8-15,24-31}
      int row = wn * 32 + j * 16 + fr;
      int kb = half * 8;
      bg[j].h[0] = *(const bf16x8*)&Bg[row * 40 + kb];
      bg[j].h[1] = *(const bf16x8*)&Bg[row * 40 + kb + 16];
      bv[j].h[0] = *(const bf16x8*)&Bv[row * 40 + kb];
      bv[j].h[1] = *(const bf16x8*)&Bv[row * 40 + kb + 16];
    }
    #pragma unroll
    for (int i = 0; i < 2; i++)
      #pragma unroll
      for (int j = 0; j < 2; j++) {
        accG[i][j] = __builtin_amdgcn_wmma_f32_16x16x32_bf16(
            false, a[i].v, false, bg[j].v, (short)0, accG[i][j], false, false);
        accV[i][j] = __builtin_amdgcn_wmma_f32_16x16x32_bf16(
            false, a[i].v, false, bv[j].v, (short)0, accV[i][j], false, false);
      }
    __syncthreads();
  }

  const int moff = half * 8;
  const int ncol = lane & 15;
  #pragma unroll
  for (int i = 0; i < 2; i++)
    #pragma unroll
    for (int j = 0; j < 2; j++) {
      int n = tileN + wn * 32 + j * 16 + ncol;
      float gbn = gb[n], vbn = vb[n];
      #pragma unroll
      for (int e = 0; e < 8; e++) {
        int m = tileM + wm * 32 + i * 16 + moff + e;
        float gl = accG[i][j][e] + gbn;
        float vl = accV[i][j][e] + vbn;
        float g  = 1.0f / (1.0f + __expf(-gl));
        U[(size_t)m * D_DIM + n] = (2.0f * g - 1.0f) * vl;
      }
    }
}

// ---------------------------------------------------------------- GEMM: out proj + residual
__global__ __launch_bounds__(256) void gemm_out_kernel(
    const __bf16* __restrict__ A, const __bf16* __restrict__ Wo,
    const float* __restrict__ ob, const float* __restrict__ Xin,
    float* __restrict__ Out) {
  __shared__ __align__(16) __bf16 As[128 * 40];
  __shared__ __align__(16) __bf16 Bs[64 * 40];
  const int tid  = threadIdx.x;
  const int lane = tid & 31, wave = tid >> 5;
  const int wm = wave & 3, wn = wave >> 2;
  const int tileM = blockIdx.x * 128;
  const int tileN = blockIdx.y * 64;
  const int fr = lane & 15;
  const int half = lane >> 4;

  const int ar0 = tid >> 2,         ac0 = (tid & 3) * 8;
  const int ar1 = (tid + 256) >> 2, ac1 = ((tid + 256) & 3) * 8;
  const int br  = tid >> 2,         bc  = (tid & 3) * 8;

  v8f acc[2][2] = {};

  for (int k0 = 0; k0 < D_DIM; k0 += 32) {
    bf16x8 tA0 = *(const bf16x8*)&A [(size_t)(tileM + ar0) * D_DIM + k0 + ac0];
    bf16x8 tA1 = *(const bf16x8*)&A [(size_t)(tileM + ar1) * D_DIM + k0 + ac1];
    bf16x8 tB  = *(const bf16x8*)&Wo[(size_t)(tileN + br ) * D_DIM + k0 + bc ];
    if (k0 + 96 < D_DIM) {
      __builtin_prefetch(&A[(size_t)(tileM + ar0) * D_DIM + k0 + 64 + ac0], 0, 1);
      __builtin_prefetch(&A[(size_t)(tileM + ar1) * D_DIM + k0 + 64 + ac1], 0, 1);
    }
    *(bf16x8*)&As[ar0 * 40 + ac0] = tA0;
    *(bf16x8*)&As[ar1 * 40 + ac1] = tA1;
    *(bf16x8*)&Bs[br  * 40 + bc ] = tB;
    __syncthreads();

    Frag a[2], bb[2];
    #pragma unroll
    for (int i = 0; i < 2; i++) {
      int row = wm * 32 + i * 16 + fr;
      a[i].h[0] = *(const bf16x8*)&As[row * 40 + half * 8];
      a[i].h[1] = *(const bf16x8*)&As[row * 40 + half * 8 + 16];
    }
    #pragma unroll
    for (int j = 0; j < 2; j++) {
      int row = wn * 32 + j * 16 + fr;
      int kb = half * 8;
      bb[j].h[0] = *(const bf16x8*)&Bs[row * 40 + kb];
      bb[j].h[1] = *(const bf16x8*)&Bs[row * 40 + kb + 16];
    }
    #pragma unroll
    for (int i = 0; i < 2; i++)
      #pragma unroll
      for (int j = 0; j < 2; j++)
        acc[i][j] = __builtin_amdgcn_wmma_f32_16x16x32_bf16(
            false, a[i].v, false, bb[j].v, (short)0, acc[i][j], false, false);
    __syncthreads();
  }

  const int moff = half * 8;
  const int ncol = lane & 15;
  #pragma unroll
  for (int i = 0; i < 2; i++)
    #pragma unroll
    for (int j = 0; j < 2; j++) {
      int n = tileN + wn * 32 + j * 16 + ncol;
      float obn = ob[n];
      #pragma unroll
      for (int e = 0; e < 8; e++) {
        int m = tileM + wm * 32 + i * 16 + moff + e;
        size_t idx = (size_t)m * D_DIM + n;
        Out[idx] = Xin[idx] + acc[i][j][e] + obn;
      }
    }
}

// ---------------------------------------------------------------- scan (two-pass, 32 chunks of 128 over T)
__global__ __launch_bounds__(256) void scan_partial_kernel(
    const float* __restrict__ U, float* __restrict__ part) {
  int g = blockIdx.x * blockDim.x + threadIdx.x;   // 0 .. 131071
  int c = g & 4095;                                // column = b*1024 + d
  int p = g >> 12;                                 // chunk 0..31
  int b = c >> 10, d = c & 1023;
  size_t base = (size_t)b * T_DIM * D_DIM + (size_t)(p * 128) * D_DIM + d;
  float s = 0.f;
  for (int t = 0; t < 128; t++) s += U[base + (size_t)t * D_DIM];
  part[p * 4096 + c] = s;
}

__global__ __launch_bounds__(256) void scan_apply_kernel(
    const float* __restrict__ U, const float* __restrict__ part,
    const float* __restrict__ log_decay, __bf16* __restrict__ S) {
  int g = blockIdx.x * blockDim.x + threadIdx.x;
  int c = g & 4095;
  int p = g >> 12;
  int b = c >> 10, d = c & 1023;
  float carry = 0.f;
  for (int q = 0; q < p; q++) carry += part[q * 4096 + c];
  float alpha = log1pf(__expf(log_decay[0]));      // softplus
  size_t base = (size_t)b * T_DIM * D_DIM + (size_t)(p * 128) * D_DIM + d;
  int t0 = p * 128;
  float run  = carry;
  float dec  = expf(-alpha * (float)t0);           // seed once per 128-chunk
  float fmul = expf(-alpha);                       // per-step decay factor
  for (int t = 0; t < 128; t++) {
    run += U[base + (size_t)t * D_DIM];
    S[base + (size_t)t * D_DIM] = f2bf(run * dec);
    dec *= fmul;
  }
}

// ---------------------------------------------------------------- launch
extern "C" void kernel_launch(void* const* d_in, const int* in_sizes, int n_in,
                              void* d_out, int out_size, void* d_ws, size_t ws_size,
                              hipStream_t stream) {
  const float* x   = (const float*)d_in[0];
  const float* lnw = (const float*)d_in[1];
  const float* lnb = (const float*)d_in[2];
  const float* gw  = (const float*)d_in[3];
  const float* gb  = (const float*)d_in[4];
  const float* vw  = (const float*)d_in[5];
  const float* vb  = (const float*)d_in[6];
  const float* ow  = (const float*)d_in[7];
  const float* ob  = (const float*)d_in[8];
  const float* ld  = (const float*)d_in[9];
  float* out = (float*)d_out;

  char* ws = (char*)d_ws;
  __bf16* normed = (__bf16*)ws; ws += (size_t)M_DIM * D_DIM * 2;   // 32 MB
  __bf16* wg_bf  = (__bf16*)ws; ws += (size_t)D_DIM * D_DIM * 2;   //  2 MB
  __bf16* wv_bf  = (__bf16*)ws; ws += (size_t)D_DIM * D_DIM * 2;   //  2 MB
  __bf16* wo_bf  = (__bf16*)ws; ws += (size_t)D_DIM * D_DIM * 2;   //  2 MB
  float*  U      = (float*)ws;  ws += (size_t)M_DIM * D_DIM * 4;   // 64 MB
  __bf16* S      = (__bf16*)ws; ws += (size_t)M_DIM * D_DIM * 2;   // 32 MB
  float*  part   = (float*)ws;                                     // 512 KB

  cvt_f32_bf16_kernel<<<1024, 256, 0, stream>>>(gw, wg_bf, D_DIM * D_DIM);
  cvt_f32_bf16_kernel<<<1024, 256, 0, stream>>>(vw, wv_bf, D_DIM * D_DIM);
  cvt_f32_bf16_kernel<<<1024, 256, 0, stream>>>(ow, wo_bf, D_DIM * D_DIM);

  ln_bf16_kernel<<<M_DIM, 256, 0, stream>>>(x, lnw, lnb, normed);

  dim3 gemm_grid(M_DIM / 128, D_DIM / 64);   // 128 x 16
  gemm_gv_kernel<<<gemm_grid, 256, 0, stream>>>(normed, wg_bf, wv_bf, gb, vb, U);

  scan_partial_kernel<<<512, 256, 0, stream>>>(U, part);
  scan_apply_kernel<<<512, 256, 0, stream>>>(U, part, ld, S);

  gemm_out_kernel<<<gemm_grid, 256, 0, stream>>>(S, wo_bf, ob, x, out);
}
